// SSTLoss_2164663517434
// MI455X (gfx1250) — compile-verified, matrix operand order checked by reading, not run on hardware
//
#include <hip/hip_runtime.h>
#include <stdint.h>

#define MP1 81          // M+1
#define MM  80          // M
#define NEL (MP1 * MP1) // 6561
#define NTHREADS 256
#define PADR 96         // rows/cols padded to 6 blocks of 16

typedef __attribute__((ext_vector_type(16))) _Float16     v16h;
typedef __attribute__((ext_vector_type(8)))  float        v8f;
typedef __attribute__((ext_vector_type(4)))  unsigned int u32x4;
typedef __attribute__((ext_vector_type(4)))  int          i32x4;
typedef __attribute__((ext_vector_type(8)))  int          i32x8;

#if defined(__gfx1250__) && __has_builtin(__builtin_amdgcn_tensor_load_to_lds)
#define USE_TDM 1
#else
#define USE_TDM 0
#endif

// Compile-time marker so stderr tells us which staging path the DEVICE pass took.
#ifdef __gfx1250__
#if USE_TDM
#warning "gfx1250 device pass: TDM tensor_load_to_lds path ENABLED"
#else
#warning "gfx1250 device pass: TDM builtin missing -> fallback vector loads"
#endif
#endif

// Deterministic 256-thread block reduction (wave32 shfl tree + fixed-order LDS).
// Result valid on thread 0 only.
__device__ __forceinline__ float blockReduceSum(float v, float* sTmp) {
#pragma unroll
    for (int off = 16; off > 0; off >>= 1) v += __shfl_xor(v, off, 32);
    const int lane = threadIdx.x & 31;
    const int w    = threadIdx.x >> 5;
    __syncthreads();                 // protect sTmp from any previous use
    if (lane == 0) sTmp[w] = v;
    __syncthreads();
    float r = 0.f;
    if (threadIdx.x == 0) {
#pragma unroll
        for (int i = 0; i < 8; i++) r += sTmp[i];
    }
    return r;
}

// One 16-row (or 16-col) exp-sum task: D = A(exp tile, f16) x ones + C.
// Every column of D holds the 16 line sums; accumulate f32 over 3 K-chunks.
// Fully branchless: indices clamped to 80 (always in-bounds; sRM/sCM[80]==0
// makes clamped lanes compute e==1) and zeroed via a 0/1 multiply.
template <bool ISROW>
__device__ __forceinline__ void expSumTask(
    int blk, int lm, int lh,
    const float* __restrict__ sIN,
    const float* __restrict__ sM0f, const float* __restrict__ sM1f,
    const float* __restrict__ sMX,   // sRM (row) / sCM (col)
    float* __restrict__ sOut,        // sRS (row) / sCS (col)
    v16h ones)
{
    const int   m      = blk * 16 + lm;
    const int   mcl    = (m < MM) ? m : MM;        // clamp line index
    const float fM     = ISROW ? sM0f[mcl] : sM1f[mcl];
    const float mx     = sMX[mcl];
    const float lf     = (m < MM) ? 1.f : 0.f;     // line 80: fully masked -> v=0
    const float mValid = (m < MP1) ? 1.f : 0.f;
    v8f acc = {0.f, 0.f, 0.f, 0.f, 0.f, 0.f, 0.f, 0.f};
#pragma unroll
    for (int kc = 0; kc < 3; kc++) {
        v16h a;
#pragma unroll
        for (int j = 0; j < 16; j++) {
            // f16 16x32 A layout: element j of lane-group lh -> K index:
            const int   k     = kc * 32 + ((j >> 3) << 4) + (lh << 3) + (j & 7);
            const int   kcl   = (k < MM) ? k : MM;
            const float fK    = ISROW ? sM1f[kcl] : sM0f[kcl];
            const float inv   = ISROW ? sIN[mcl * MP1 + kcl] : sIN[kcl * MP1 + mcl];
            const float valid = ((k < MP1) ? 1.f : 0.f) * mValid;
            const float v     = lf * fM * fK * inv;
            const float e     = __expf(v - mx) * valid;
            a[j] = (_Float16)e;
        }
        acc = __builtin_amdgcn_wmma_f32_16x16x32_f16(false, a, false, ones,
                                                     (short)0, acc, false, false);
    }
    // D layout: lanes 0-15 VGPR i -> M=i ; lanes 16-31 VGPR i -> M=8+i.
    if (lm == 0) {
#pragma unroll
        for (int i = 0; i < 8; i++) {
            const int mm = blk * 16 + (lh << 3) + i;
            if (mm < MP1) sOut[mm] = acc[i];
        }
    }
}

__global__ __launch_bounds__(NTHREADS)
void sst_main(const float* __restrict__ input, const int* __restrict__ target,
              const int* __restrict__ mask0, const int* __restrict__ mask1,
              float* __restrict__ idxOut,   // d_out + 7, [B][80] argmax indices (as float)
              float* __restrict__ ws)       // [B][12] block partials
{
    __shared__ float sIN[NEL];
    __shared__ float sT[NEL];
    __shared__ float sM0f[PADR], sM1f[PADR];
    __shared__ float sRM[PADR], sCM[PADR];      // row/col max of masked values
    __shared__ float sRS[PADR], sCS[PADR];      // row/col sums of exp (WMMA)
    __shared__ float sLRS[PADR], sLCS[PADR];    // logs of sums
    __shared__ float sTmp[8];

    const int tid  = threadIdx.x;
    const int b    = blockIdx.x;
    const int wid  = tid >> 5;
    const int lane = tid & 31;
    const size_t base = (size_t)b * NEL;

    // ---------------- Stage tiles into LDS ----------------
#if USE_TDM
    if (wid == 0) {
        // Tensor DMA descriptor: 1-D tile of 6561 x 4B, contiguous.
        const uint32_t ldsOff = (uint32_t)(uintptr_t)(void*)&sIN[0];
        const uint64_t ga     = (uint64_t)(uintptr_t)(input + base);
        u32x4 g0;
        g0.x = 1u;                                   // count=1 (user mode)
        g0.y = ldsOff;                               // lds_addr (bytes)
        g0.z = (uint32_t)ga;                         // global_addr[31:0]
        g0.w = (uint32_t)(ga >> 32) | (2u << 30);    // global_addr[56:32] | type=2
        i32x8 g1;
        g1[0] = 0x00020000;                          // data_size = 4B
        g1[1] = (int)(((unsigned)NEL & 0xFFFFu) << 16);      // tensor_dim0[15:0]
        g1[2] = (int)(((unsigned)NEL >> 16) | (1u << 16));   // tensor_dim0 hi | tensor_dim1=1 lo
        g1[3] = (int)((unsigned)NEL << 16);                  // tensor_dim1 hi=0 | tile_dim0=NEL
        g1[4] = 1;                                           // tile_dim1=1, tile_dim2=0
        g1[5] = NEL;                                         // tensor_dim0_stride lo32
        g1[6] = 0; g1[7] = 0;
        i32x4 z4 = {0, 0, 0, 0};
#if __clang_major__ >= 23
        i32x8 z8 = {0, 0, 0, 0, 0, 0, 0, 0};
        __builtin_amdgcn_tensor_load_to_lds(g0, g1, z4, z4, z8, 0);
#else
        __builtin_amdgcn_tensor_load_to_lds(g0, g1, z4, z4, 0);
#endif
        __builtin_amdgcn_s_wait_tensorcnt(0);
    }
#else
    for (int i = tid; i < NEL; i += NTHREADS)
        sIN[i] = __builtin_nontemporal_load(input + base + i);
#endif
    // Target (int -> 0/1 float) and masks, loaded by all waves (overlaps the TDM).
    for (int i = tid; i < NEL; i += NTHREADS)
        sT[i] = __builtin_nontemporal_load(target + base + i) ? 1.f : 0.f;
    if (tid < MP1) {
        sM0f[tid] = (float)mask0[(size_t)b * MP1 + tid];
        sM1f[tid] = (float)mask1[(size_t)b * MP1 + tid];
    } else if (tid < PADR) {
        sM0f[tid] = 0.f; sM1f[tid] = 0.f;
    }
    __syncthreads();

    // ---------------- Phase A: row/col maxima of masked values ----------------
    // pre:  v[r][c] = (r==M) ? 0 : m0[r]*m1[c]*in[r][c]   (softmax over c)
    // next: v[r][c] = (c==M) ? 0 : m0[r]*m1[c]*in[r][c]   (softmax over r)
    if (tid < MP1) {
        const int t = tid;
        const float alive = (t == MM) ? 0.f : 1.f; // row M dead (pre); col M dead (next)
        float rm = -3.0e38f, cm = -3.0e38f;
        for (int k = 0; k < MP1; k++) {
            const float vp = alive * sM0f[t] * sM1f[k] * sIN[t * MP1 + k];
            rm = fmaxf(rm, vp);
            const float vn = alive * sM0f[k] * sM1f[t] * sIN[k * MP1 + t];
            cm = fmaxf(cm, vn);
        }
        sRM[t] = rm; sCM[t] = cm;
    } else if (tid < PADR) {
        sRM[tid] = 0.f; sCM[tid] = 0.f; sRS[tid] = 1.f; sCS[tid] = 1.f;
    }
    __syncthreads();

    // ---------------- Phase B: exp-sum rows/cols via WMMA -------------------
    v16h ones;
#pragma unroll
    for (int j = 0; j < 16; j++) ones[j] = (_Float16)1.f;

    const int lm = lane & 15;
    const int lh = lane >> 4;
    for (int task = wid; task < 12; task += 8) {     // wave-uniform -> EXEC all 1s
        if (task < 6) expSumTask<true >(task,     lm, lh, sIN, sM0f, sM1f, sRM, sRS, ones);
        else          expSumTask<false>(task - 6, lm, lh, sIN, sM0f, sM1f, sCM, sCS, ones);
    }
    __syncthreads();

    if (tid < MP1) { sLRS[tid] = __logf(sRS[tid]); sLCS[tid] = __logf(sCS[tid]); }
    __syncthreads();

    // ---------------- Phase C: element-wise sums ----------------------------
    float a_tnum = 0.f, a_npre = 0.f, a_nnext = 0.f, a_nun = 0.f;
    float a_spre = 0.f, a_snext = 0.f, a_sall = 0.f, a_ssim = 0.f;
    for (int i = tid; i < NEL; i += NTHREADS) {
        const int r = i / MP1;
        const int c = i - r * MP1;
        const float in = sIN[i], t = sT[i];
        const float mreg = sM0f[r] * sM1f[c];
        const float mp = (r == MM) ? 0.f : mreg;
        const float mn = (c == MM) ? 0.f : mreg;
        const float mu = ((r == MM) || (c == MM)) ? 0.f : mreg;
        const float lp  = (mp * in - sRM[r]) - sLRS[r];   // log(input_pre)
        const float lnx = (mn * in - sCM[c]) - sLCS[c];   // log(input_next)
        const float pre = __expf(lp);
        const float nxt = __expf(lnx);
        const float tpre = mp * t, tnext = mn * t, tun = mu * t;
        a_tnum  += t;
        a_npre  += tpre;  a_nnext += tnext;  a_nun += tun;
        a_spre  -= tpre * lp;
        a_snext -= tnext * lnx;
        const float lall = (r < MM && c < MM) ? fmaxf(lp, lnx) : lp; // log(max)=max(log)
        a_sall  -= tpre * lall;
        a_ssim  += tun * fabsf(nxt - pre);
    }

    float red[12];
    red[0] = blockReduceSum(a_tnum, sTmp);
    red[1] = blockReduceSum(a_npre, sTmp);
    red[2] = blockReduceSum(a_nnext, sTmp);
    red[3] = blockReduceSum(a_nun, sTmp);
    red[4] = blockReduceSum(a_spre, sTmp);
    red[5] = blockReduceSum(a_snext, sTmp);
    red[6] = blockReduceSum(a_sall, sTmp);
    red[7] = blockReduceSum(a_ssim, sTmp);

    // ---------------- Phase D: row argmax (pre/all) + accuracy --------------
    float accp = 0.f, nmp = 0.f;
    if (tid < MM) {
        const int r = tid;                     // r < 80
        float best = -3.0e38f; int bi = 0;
        float bt   = -3.0e38f; int bti = 0;
        for (int c = 0; c < MP1; c++) {
            const float in = sIN[r * MP1 + c], t = sT[r * MP1 + c];
            const float mreg = sM0f[r] * sM1f[c];
            const float mn   = (c == MM) ? 0.f : mreg;
            const float pre  = __expf(mreg * in - sRM[r] - sLRS[r]);
            const float nxt  = __expf(mn * in - sCM[c] - sLCS[c]);
            const float all  = (c < MM) ? fmaxf(pre, nxt) : pre;
            if (all > best) { best = all; bi = c; }       // first-occurrence argmax
            const float tp = mreg * t;
            if (tp > bt) { bt = tp; bti = c; }
        }
        accp = (bi == bti) ? sM0f[r] : 0.f;
        nmp  = sM0f[r];
        __builtin_nontemporal_store((float)bi, idxOut + (size_t)b * MM + r);
    }
    red[8] = blockReduceSum(accp, sTmp);
    red[9] = blockReduceSum(nmp, sTmp);

    // ---------------- Phase E: col argmax (next) + accuracy -----------------
    float accn = 0.f, nmn = 0.f;
    if (tid < MM) {
        const int c = tid;                     // c < 80
        float best = -3.0e38f; int bi = 0;
        float bt   = -3.0e38f; int bti = 0;
        for (int r = 0; r < MP1; r++) {
            const float in = sIN[r * MP1 + c], t = sT[r * MP1 + c];
            const float mn = sM0f[r] * sM1f[c];          // c < 80 -> not zeroed
            const float nxt = __expf(mn * in - sCM[c] - sLCS[c]);
            if (nxt > best) { best = nxt; bi = r; }
            const float tn = mn * t;
            if (tn > bt) { bt = tn; bti = r; }
        }
        accn = (bi == bti) ? sM1f[c] : 0.f;
        nmn  = sM1f[c];
    }
    red[10] = blockReduceSum(accn, sTmp);
    red[11] = blockReduceSum(nmn, sTmp);

    if (tid == 0) {
#pragma unroll
        for (int k = 0; k < 12; k++) ws[(size_t)b * 12 + k] = red[k];
    }
}

// Deterministic cross-block reduction + scalar epilogue.
__global__ __launch_bounds__(NTHREADS)
void sst_finalize(const float* __restrict__ ws, float* __restrict__ out, int B)
{
    __shared__ float sTmp[8];
    __shared__ float sums[12];
    const int tid = threadIdx.x;
    for (int k = 0; k < 12; k++) {
        float v = 0.f;
        for (int i = tid; i < B; i += NTHREADS) v += ws[(size_t)i * 12 + k];
#pragma unroll
        for (int off = 16; off > 0; off >>= 1) v += __shfl_xor(v, off, 32);
        __syncthreads();
        if ((tid & 31) == 0) sTmp[tid >> 5] = v;
        __syncthreads();
        if (tid == 0) { float s = 0.f; for (int i = 0; i < 8; i++) s += sTmp[i]; sums[k] = s; }
        __syncthreads();
    }
    if (tid == 0) {
        const float t_num = sums[0], n_pre = sums[1], n_next = sums[2], n_un = sums[3];
        const float s_pre = sums[4], s_next = sums[5], s_all = sums[6], s_sim = sums[7];
        const float accp = sums[8], nmp = sums[9], accn = sums[10], nmn = sums[11];
        const float loss_pre  = (n_pre  > 0.f) ? s_pre  / n_pre  : s_pre;
        const float loss_next = (n_next > 0.f) ? s_next / n_next : s_next;
        const float loss      = (n_pre > 0.f && n_next > 0.f) ? s_all / n_pre : s_all;
        const float loss_sim  = (n_un > 0.f) ? s_sim / t_num : s_sim;
        const float acc_pre   = (nmp > 0.f) ? accp / nmp : accp + 1.f;
        const float acc_next  = (nmn > 0.f) ? accn / nmn : accn + 1.f;
        out[0] = loss_pre;
        out[1] = loss_next;
        out[2] = loss_sim;
        out[3] = (loss_pre + loss_next + loss + loss_sim) * 0.25f;
        out[4] = acc_pre;
        out[5] = acc_next;
        out[6] = (acc_pre + acc_next) * 0.5f;
    }
}

extern "C" void kernel_launch(void* const* d_in, const int* in_sizes, int n_in,
                              void* d_out, int out_size, void* d_ws, size_t ws_size,
                              hipStream_t stream) {
    (void)n_in; (void)out_size; (void)ws_size;
    const float* input  = (const float*)d_in[0];
    const int*   target = (const int*)d_in[1];
    const int*   mask0  = (const int*)d_in[2];
    const int*   mask1  = (const int*)d_in[3];
    float* out = (float*)d_out;
    float* ws  = (float*)d_ws;
    const int B = in_sizes[0] / NEL;   // 2048

    sst_main<<<B, NTHREADS, 0, stream>>>(input, target, mask0, mask1, out + 7, ws);
    sst_finalize<<<1, NTHREADS, 0, stream>>>(ws, out, B);
}
